// BasicTransformerBlock_87376814670426
// MI455X (gfx1250) — compile-verified, matrix operand order checked by reading, not run on hardware
//
#include <hip/hip_runtime.h>
#include <math.h>

// ---------------- problem constants ----------------
#define DIM_      1024
#define CTX_DIM_  768
#define HEADS_    16
#define DHEAD_    64
#define INNER_    1024
#define FF_INNER_ 4096
#define RANK_     16
#define BATCH_    2
#define NSEQ_     2048
#define NCTX_     77
#define ROWS_X    (BATCH_ * NSEQ_)   // 4096
#define ROWS_C    (BATCH_ * NCTX_)   // 154

typedef __attribute__((ext_vector_type(16))) _Float16 v16h;
typedef __attribute__((ext_vector_type(8)))  float    v8f;

union Frag16 { v16h  v; uint4 q[2]; };
union H4     { _Float16 h[4]; uint2 u; };

__device__ inline v8f wmma_ff(v16h a, v16h b, v8f c) {
  // D = A(16x32 f16) x B(32x16 f16) + C(16x16 f32)
  return __builtin_amdgcn_wmma_f32_16x16x32_f16(false, a, false, b, (short)0, c,
                                                false, false);
}

// ---- CDNA5 async global->LDS staging (ASYNCcnt-tracked, bypasses VGPRs) ----
#define AS3 __attribute__((address_space(3)))
__device__ inline unsigned lds_off(const void* p) {
  // generic -> LDS addrspace cast yields the raw 32-bit LDS byte offset
  return (unsigned)(unsigned long long)(AS3 const void*)p;
}
__device__ inline void async_ld_b128(unsigned lds_dst, const void* gsrc) {
  asm volatile("global_load_async_to_lds_b128 %0, %1, off"
               :: "v"(lds_dst), "v"(gsrc) : "memory");
}
__device__ inline void wait_async0() {
  asm volatile("s_wait_asynccnt 0x0" ::: "memory");
}

// A/B fragment loader: row32 points at 32 contiguous f16 (k=0..31) belonging to
// this lane's matrix row (A: M = lane&15, B stored transposed: N = lane&15).
// ISA 16-bit A layout: lanes<16 hold K {0..7,16..23}, lanes>=16 hold {8..15,24..31}.
__device__ inline v16h load_frag32(const _Float16* row32) {
  const int kb = (threadIdx.x & 16) ? 8 : 0;
  Frag16 f;
  f.q[0] = *(const uint4*)(row32 + kb);
  f.q[1] = *(const uint4*)(row32 + kb + 16);
  return f.v;
}

// gather version for LDS tiles stored [k][n] with row stride `stride`
__device__ inline v16h load_frag_gather(const _Float16* lds, int stride, int col) {
  const int kb = (threadIdx.x & 16) ? 8 : 0;
  v16h f;
#pragma unroll
  for (int e = 0; e < 16; ++e) {
    int k = kb + e + ((e & 8) ? 8 : 0);
    f[e] = lds[k * stride + col];
  }
  return f;
}

// ---------------- LayerNorm -> f16 ----------------
__global__ __launch_bounds__(256)
void ln_f16_kernel(const float* __restrict__ x, const float* __restrict__ g,
                   const float* __restrict__ b, _Float16* __restrict__ out) {
  __shared__ float red[256];
  const int r = blockIdx.x, tid = threadIdx.x;
  const float* xr = x + (size_t)r * DIM_;
  float4 v = *(const float4*)(xr + tid * 4);
  red[tid] = v.x + v.y + v.z + v.w;
  __syncthreads();
  for (int o = 128; o > 0; o >>= 1) {
    if (tid < o) red[tid] += red[tid + o];
    __syncthreads();
  }
  const float mean = red[0] * (1.0f / DIM_);
  __syncthreads();
  float d0 = v.x - mean, d1 = v.y - mean, d2 = v.z - mean, d3 = v.w - mean;
  red[tid] = d0 * d0 + d1 * d1 + d2 * d2 + d3 * d3;
  __syncthreads();
  for (int o = 128; o > 0; o >>= 1) {
    if (tid < o) red[tid] += red[tid + o];
    __syncthreads();
  }
  const float rstd = rsqrtf(red[0] * (1.0f / DIM_) + 1e-5f);
  float4 gg = *(const float4*)(g + tid * 4);
  float4 bb = *(const float4*)(b + tid * 4);
  _Float16* o4 = out + (size_t)r * DIM_ + tid * 4;
  o4[0] = (_Float16)(d0 * rstd * gg.x + bb.x);
  o4[1] = (_Float16)(d1 * rstd * gg.y + bb.y);
  o4[2] = (_Float16)(d2 * rstd * gg.z + bb.z);
  o4[3] = (_Float16)(d3 * rstd * gg.w + bb.w);
}

// ---------------- f32 -> f16 convert ----------------
__global__ __launch_bounds__(256)
void cvt_f16_kernel(const float* __restrict__ src, _Float16* __restrict__ dst, int n) {
  int i = blockIdx.x * 256 + threadIdx.x;
  if (i < n) dst[i] = (_Float16)src[i];
}

// ---------------- LoRA down: T = (A @ down) * (alpha/rank) ----------------
__global__ __launch_bounds__(256)
void lora_down_kernel(const _Float16* __restrict__ A, int M, int K,
                      const float* __restrict__ down, const float* __restrict__ alpha,
                      float* __restrict__ T) {
  int idx = blockIdx.x * 256 + threadIdx.x;
  if (idx >= M * RANK_) return;
  int r = idx >> 4, j = idx & 15;
  const _Float16* ar = A + (size_t)r * K;
  float s = 0.0f;
  for (int k = 0; k < K; ++k) s += (float)ar[k] * down[k * RANK_ + j];
  T[idx] = s * (alpha[0] * (1.0f / RANK_));   // LORA_W == 1
}

// ---------------- main WMMA GEMM ----------------
// C(MxN f32) = A(MxK f16) @ B(KxN f16) [+ bias] [+ loraT(Mx16) @ loraU(16xN)]
// MODE 0: write f16 to outH.  MODE 1: resid[idx] += C (residual accumulate).
template <int MODE>
__global__ __launch_bounds__(256)
void gemm_wmma_kernel(const _Float16* __restrict__ A, const _Float16* __restrict__ Bw,
                      int M, int K, int N,
                      const float* __restrict__ bias,
                      const float* __restrict__ loraT, const float* __restrict__ loraU,
                      _Float16* __restrict__ outH, float* __restrict__ resid) {
  __shared__ __align__(16) _Float16 As[128 * 32];
  __shared__ __align__(16) _Float16 Bs[64 * 32];   // stored transposed: [n][k]
  const int tid = threadIdx.x, lane = tid & 31, wid = tid >> 5;
  const int m0 = blockIdx.y * 128, n0 = blockIdx.x * 64;
  const int wm = (wid >> 1) * 32, wn = (wid & 1) * 32;
  const v8f vz = {};
  v8f acc[2][2];
  for (int i = 0; i < 2; ++i)
    for (int j = 0; j < 2; ++j) acc[i][j] = vz;

  const int nk = K / 32;
  for (int kt = 0; kt < nk; ++kt) {
    const int k0 = kt * 32;
    // A tile: pure copy -> CDNA5 async global->LDS (no VGPR round-trip).
    // OOB rows (cross-attn M=154) just skip: their LDS garbage only feeds
    // output rows that the epilogue discards.
#pragma unroll
    for (int i = 0; i < 2; ++i) {
      int ch = tid * 2 + i;
      int r = ch >> 2, kg = (ch & 3) * 8;
      int gr = m0 + r;
      if (gr < M)
        async_ld_b128(lds_off(As + r * 32 + kg), A + (size_t)gr * K + k0 + kg);
    }
    {  // B tile: transpose-stage, 4(k) x 2(n) patch per thread -> 2 b64 LDS stores
      int n = (lane) * 2 + 0;          // placeholder, real mapping below
      (void)n;
      int np = (tid & 31) * 2, kk = (tid >> 5) * 4;
      H4 lo, hi;
#pragma unroll
      for (int j = 0; j < 4; ++j) {
        unsigned rv = *(const unsigned*)(Bw + (size_t)(k0 + kk + j) * N + n0 + np);
        const _Float16* hp = (const _Float16*)&rv;
        lo.h[j] = hp[0];
        hi.h[j] = hp[1];
      }
      if (kt + 1 < nk)
        __builtin_prefetch(Bw + (size_t)(k0 + 32 + kk) * N + n0 + np, 0, 1);
      *(uint2*)(Bs + (np + 0) * 32 + kk) = lo.u;
      *(uint2*)(Bs + (np + 1) * 32 + kk) = hi.u;
    }
    wait_async0();
    __syncthreads();
    v16h af0 = load_frag32(As + (wm + (lane & 15)) * 32);
    v16h af1 = load_frag32(As + (wm + 16 + (lane & 15)) * 32);
    v16h bf0 = load_frag32(Bs + (wn + (lane & 15)) * 32);
    v16h bf1 = load_frag32(Bs + (wn + 16 + (lane & 15)) * 32);
    acc[0][0] = wmma_ff(af0, bf0, acc[0][0]);
    acc[0][1] = wmma_ff(af0, bf1, acc[0][1]);
    acc[1][0] = wmma_ff(af1, bf0, acc[1][0]);
    acc[1][1] = wmma_ff(af1, bf1, acc[1][1]);
    __syncthreads();
  }

  const int lh = (lane >> 4) & 1, nl = lane & 15;
  for (int ni = 0; ni < 2; ++ni) {
    const int col = n0 + wn + ni * 16 + nl;
    float u[16];
    if (loraU) {
#pragma unroll
      for (int j = 0; j < 16; ++j) u[j] = loraU[j * N + col];
    }
    const float bv = bias ? bias[col] : 0.0f;
    for (int mi = 0; mi < 2; ++mi) {
#pragma unroll
      for (int e = 0; e < 8; ++e) {
        const int row = m0 + wm + mi * 16 + e + 8 * lh;
        if (row >= M) continue;
        float val = acc[mi][ni][e] + bv;
        if (loraU) {
          const float* tr = loraT + (size_t)row * 16;
          float s = 0.0f;
#pragma unroll
          for (int j = 0; j < 16; ++j) s += tr[j] * u[j];
          val += s;
        }
        size_t idx = (size_t)row * N + col;
        if (MODE == 0) outH[idx] = (_Float16)val;
        else           resid[idx] += val;
      }
    }
  }
}

// ---------------- GEGLU GEMM: h = (x@Wp_a + ...) * gelu(x@Wp_g + ...) ----------------
__global__ __launch_bounds__(256)
void geglu_wmma_kernel(const _Float16* __restrict__ A, const _Float16* __restrict__ Wp,
                       const float* __restrict__ bias, const float* __restrict__ loraT,
                       const float* __restrict__ loraU, _Float16* __restrict__ outH) {
  __shared__ __align__(16) _Float16 As[128 * 32];
  __shared__ __align__(16) _Float16 Bsa[64 * 32];
  __shared__ __align__(16) _Float16 Bsg[64 * 32];
  const int K = DIM_, NW = 2 * FF_INNER_;
  const int tid = threadIdx.x, lane = tid & 31, wid = tid >> 5;
  const int m0 = blockIdx.y * 128, n0 = blockIdx.x * 64;
  const int wm = (wid >> 1) * 32, wn = (wid & 1) * 32;
  const v8f vz = {};
  v8f accA[2][2], accG[2][2];
  for (int i = 0; i < 2; ++i)
    for (int j = 0; j < 2; ++j) { accA[i][j] = vz; accG[i][j] = vz; }

  for (int kt = 0; kt < K / 32; ++kt) {
    const int k0 = kt * 32;
#pragma unroll
    for (int i = 0; i < 2; ++i) {    // A tile via async copy (always in-range)
      int ch = tid * 2 + i;
      int r = ch >> 2, kg = (ch & 3) * 8;
      async_ld_b128(lds_off(As + r * 32 + kg), A + (size_t)(m0 + r) * K + k0 + kg);
    }
    {  // both Wp halves, transpose-staged
      int np = (tid & 31) * 2, kk = (tid >> 5) * 4;
      H4 al_, ah_, gl_, gh_;
#pragma unroll
      for (int j = 0; j < 4; ++j) {
        unsigned ra = *(const unsigned*)(Wp + (size_t)(k0 + kk + j) * NW + n0 + np);
        unsigned rg = *(const unsigned*)(Wp + (size_t)(k0 + kk + j) * NW + n0 + np + FF_INNER_);
        const _Float16* pa_ = (const _Float16*)&ra;
        const _Float16* pg_ = (const _Float16*)&rg;
        al_.h[j] = pa_[0]; ah_.h[j] = pa_[1];
        gl_.h[j] = pg_[0]; gh_.h[j] = pg_[1];
      }
      *(uint2*)(Bsa + (np + 0) * 32 + kk) = al_.u;
      *(uint2*)(Bsa + (np + 1) * 32 + kk) = ah_.u;
      *(uint2*)(Bsg + (np + 0) * 32 + kk) = gl_.u;
      *(uint2*)(Bsg + (np + 1) * 32 + kk) = gh_.u;
    }
    wait_async0();
    __syncthreads();
    v16h af0 = load_frag32(As + (wm + (lane & 15)) * 32);
    v16h af1 = load_frag32(As + (wm + 16 + (lane & 15)) * 32);
    v16h ba0 = load_frag32(Bsa + (wn + (lane & 15)) * 32);
    v16h ba1 = load_frag32(Bsa + (wn + 16 + (lane & 15)) * 32);
    v16h bg0 = load_frag32(Bsg + (wn + (lane & 15)) * 32);
    v16h bg1 = load_frag32(Bsg + (wn + 16 + (lane & 15)) * 32);
    accA[0][0] = wmma_ff(af0, ba0, accA[0][0]);
    accA[0][1] = wmma_ff(af0, ba1, accA[0][1]);
    accA[1][0] = wmma_ff(af1, ba0, accA[1][0]);
    accA[1][1] = wmma_ff(af1, ba1, accA[1][1]);
    accG[0][0] = wmma_ff(af0, bg0, accG[0][0]);
    accG[0][1] = wmma_ff(af0, bg1, accG[0][1]);
    accG[1][0] = wmma_ff(af1, bg0, accG[1][0]);
    accG[1][1] = wmma_ff(af1, bg1, accG[1][1]);
    __syncthreads();
  }

  const int lh = (lane >> 4) & 1, nl = lane & 15;
  for (int ni = 0; ni < 2; ++ni) {
    const int col = n0 + wn + ni * 16 + nl;
    float ua[16], ug[16];
#pragma unroll
    for (int j = 0; j < 16; ++j) {
      ua[j] = loraU[j * NW + col];
      ug[j] = loraU[j * NW + col + FF_INNER_];
    }
    const float ba = bias[col], bg = bias[col + FF_INNER_];
    for (int mi = 0; mi < 2; ++mi) {
#pragma unroll
      for (int e = 0; e < 8; ++e) {
        const int row = m0 + wm + mi * 16 + e + 8 * lh;
        const float* tr = loraT + (size_t)row * 16;
        float sa = 0.0f, sg = 0.0f;
#pragma unroll
        for (int j = 0; j < 16; ++j) { sa += tr[j] * ua[j]; sg += tr[j] * ug[j]; }
        float av = accA[mi][ni][e] + ba + sa;
        float gv = accG[mi][ni][e] + bg + sg;
        float gel = 0.5f * gv * (1.0f + erff(gv * 0.70710678118654752f));
        outH[(size_t)row * FF_INNER_ + col] = (_Float16)(av * gel);
      }
    }
  }
}

// ---------------- flash attention (one wave per 16 queries per (b,h)) ----------------
__global__ __launch_bounds__(32)
void attn_flash_kernel(const _Float16* __restrict__ Q, const _Float16* __restrict__ Kh,
                       const _Float16* __restrict__ Vh, _Float16* __restrict__ O,
                       int Lk) {
  __shared__ __align__(16) _Float16 Qs[16 * 64];
  __shared__ __align__(16) _Float16 Ks[32 * 64];
  __shared__ __align__(16) _Float16 Vs[32 * 64];
  __shared__ __align__(16) _Float16 Ps[16 * 32];
  const int lane = threadIdx.x;
  const int lh = (lane >> 4) & 1, nl = lane & 15;
  const int qr0 = blockIdx.z * NSEQ_ + blockIdx.x * 16;
  const int hc = blockIdx.y * DHEAD_;

#pragma unroll
  for (int i = 0; i < 4; ++i) {                 // stage Q tile (16x64) async
    int ch = lane * 4 + i;
    int r = ch >> 3, g = (ch & 7) * 8;
    async_ld_b128(lds_off(Qs + r * 64 + g), Q + (size_t)(qr0 + r) * INNER_ + hc + g);
  }
  wait_async0();
  __syncthreads();
  v16h aQ0 = load_frag32(Qs + nl * 64);
  v16h aQ1 = load_frag32(Qs + nl * 64 + 32);

  float m[8], l[8];
  v8f o[4];
  const v8f vz = {};
#pragma unroll
  for (int e = 0; e < 8; ++e) { m[e] = -1e30f; l[e] = 0.0f; }
#pragma unroll
  for (int n = 0; n < 4; ++n) o[n] = vz;

  for (int kc = 0; kc < Lk; kc += 32) {
    if (kc + lane < Lk) {                       // stage K,V rows (async copy)
#pragma unroll
      for (int i = 0; i < 8; ++i) {
        int g = i * 8;
        size_t ofs = (size_t)(blockIdx.z * Lk + kc + lane) * INNER_ + hc + g;
        async_ld_b128(lds_off(Ks + lane * 64 + g), Kh + ofs);
        async_ld_b128(lds_off(Vs + lane * 64 + g), Vh + ofs);
      }
    } else {                                    // masked keys: V must be 0 (0*NaN=NaN)
      uint4 z = make_uint4(0u, 0u, 0u, 0u);
#pragma unroll
      for (int i = 0; i < 8; ++i) {
        int g = i * 8;
        *(uint4*)(Ks + lane * 64 + g) = z;
        *(uint4*)(Vs + lane * 64 + g) = z;
      }
    }
    wait_async0();
    __syncthreads();

    // S = Q @ K^T * scale : two 16x16 tiles (keys split 16/16), K-dim = dhead 64
    v8f s0 = vz, s1 = vz;
    s0 = wmma_ff(aQ0, load_frag32(Ks + nl * 64), s0);
    s0 = wmma_ff(aQ1, load_frag32(Ks + nl * 64 + 32), s0);
    s1 = wmma_ff(aQ0, load_frag32(Ks + (16 + nl) * 64), s1);
    s1 = wmma_ff(aQ1, load_frag32(Ks + (16 + nl) * 64 + 32), s1);
    s0 = s0 * 0.125f;                           // DHEAD^-0.5
    s1 = s1 * 0.125f;
    if (kc + nl >= Lk) {
#pragma unroll
      for (int e = 0; e < 8; ++e) s0[e] = -1e30f;
    }
    if (kc + 16 + nl >= Lk) {
#pragma unroll
      for (int e = 0; e < 8; ++e) s1[e] = -1e30f;
    }

    // online softmax update per row (row e+8*lh lives across the 16 lanes of a half)
#pragma unroll
    for (int e = 0; e < 8; ++e) {
      float rm = fmaxf(s0[e], s1[e]);
#pragma unroll
      for (int msk = 8; msk >= 1; msk >>= 1) rm = fmaxf(rm, __shfl_xor(rm, msk, 32));
      float mnew = fmaxf(m[e], rm);
      float al = __expf(m[e] - mnew);
      float p0 = __expf(s0[e] - mnew);
      float p1 = __expf(s1[e] - mnew);
      float rs = p0 + p1;
#pragma unroll
      for (int msk = 8; msk >= 1; msk >>= 1) rs += __shfl_xor(rs, msk, 32);
      l[e] = l[e] * al + rs;
      m[e] = mnew;
#pragma unroll
      for (int n = 0; n < 4; ++n) o[n][e] = o[n][e] * al;
      const int rowM = e + 8 * lh;
      Ps[rowM * 32 + nl] = (_Float16)p0;
      Ps[rowM * 32 + 16 + nl] = (_Float16)p1;
    }
    __syncthreads();

    // O += P(16x32) @ V(32x64)
    v16h aP = load_frag32(Ps + nl * 32);
#pragma unroll
    for (int n = 0; n < 4; ++n)
      o[n] = wmma_ff(aP, load_frag_gather(Vs, 64, n * 16 + nl), o[n]);
    __syncthreads();
  }

#pragma unroll
  for (int e = 0; e < 8; ++e) {
    const float inv = 1.0f / l[e];
    const int row = qr0 + e + 8 * lh;
#pragma unroll
    for (int n = 0; n < 4; ++n)
      O[(size_t)row * INNER_ + hc + n * 16 + nl] = (_Float16)(o[n][e] * inv);
  }
}

// ---------------- host-side parameter mapping ----------------
struct AttnP {
  const float *Wq, *Wk, *Wv, *Wo, *bo, *qd, *qu, *qa, *kd, *ku, *ka,
              *vd, *vu, *va, *od, *ou, *oa;
};
struct FFP {
  const float *Wp, *bp, *pd, *pu, *pa, *W2, *b2, *fd, *fu, *fa;
};
struct Params {
  const float *x, *context, *ln1_g, *ln1_b, *ln2_g, *ln2_b, *ln3_g, *ln3_b;
  AttnP a1, a2;
  FFP ff;
};

extern "C" void kernel_launch(void* const* d_in, const int* in_sizes, int n_in,
                              void* d_out, int out_size, void* d_ws, size_t ws_size,
                              hipStream_t stream) {
  (void)n_in; (void)out_size; (void)ws_size;
  auto F = [&](int i) { return (const float*)d_in[i]; };
  Params p;
  const bool insertion = (in_sizes[0] == ROWS_X * DIM_);
  if (insertion) {
    // file/dict insertion order, depth-first
    p.x = F(0); p.context = F(1);
    p.ln1_g = F(2); p.ln1_b = F(3); p.ln2_g = F(4); p.ln2_b = F(5);
    p.ln3_g = F(6); p.ln3_b = F(7);
    auto fa = [&](AttnP& a, int i) {
      a.Wq = F(i + 0); a.Wk = F(i + 1); a.Wv = F(i + 2); a.Wo = F(i + 3); a.bo = F(i + 4);
      a.qd = F(i + 5); a.qu = F(i + 6); a.qa = F(i + 7);
      a.kd = F(i + 8); a.ku = F(i + 9); a.ka = F(i + 10);
      a.vd = F(i + 11); a.vu = F(i + 12); a.va = F(i + 13);
      a.od = F(i + 14); a.ou = F(i + 15); a.oa = F(i + 16);
    };
    fa(p.a1, 8); fa(p.a2, 25);
    int i = 42;
    p.ff.Wp = F(i + 0); p.ff.bp = F(i + 1); p.ff.pd = F(i + 2); p.ff.pu = F(i + 3);
    p.ff.pa = F(i + 4); p.ff.W2 = F(i + 5); p.ff.b2 = F(i + 6); p.ff.fd = F(i + 7);
    p.ff.fu = F(i + 8); p.ff.fa = F(i + 9);
  } else {
    // jax tree_leaves sorted-key order
    p.context = F(0);
    auto fa = [&](AttnP& a, int i) {
      a.Wk = F(i + 0); a.Wo = F(i + 1); a.Wq = F(i + 2); a.Wv = F(i + 3); a.bo = F(i + 4);
      a.ka = F(i + 5); a.kd = F(i + 6); a.ku = F(i + 7);
      a.oa = F(i + 8); a.od = F(i + 9); a.ou = F(i + 10);
      a.qa = F(i + 11); a.qd = F(i + 12); a.qu = F(i + 13);
      a.va = F(i + 14); a.vd = F(i + 15); a.vu = F(i + 16);
    };
    fa(p.a1, 1); fa(p.a2, 18);
    int i = 35;
    p.ff.W2 = F(i + 0); p.ff.Wp = F(i + 1); p.ff.b2 = F(i + 2); p.ff.bp = F(i + 3);
    p.ff.fa = F(i + 4); p.ff.fd = F(i + 5); p.ff.fu = F(i + 6); p.ff.pa = F(i + 7);
    p.ff.pd = F(i + 8); p.ff.pu = F(i + 9);
    p.ln1_b = F(45); p.ln1_g = F(46); p.ln2_b = F(47); p.ln2_g = F(48);
    p.ln3_b = F(49); p.ln3_g = F(50);
    p.x = F(51);
  }

  // ---------------- workspace layout ----------------
  char* ws = (char*)d_ws;
  size_t off = 0;
  auto salloc = [&](size_t bytes) {
    void* q = ws + off;
    off += (bytes + 255) & ~(size_t)255;
    return q;
  };
  _Float16* wbuf = (_Float16*)salloc((size_t)DIM_ * 2 * FF_INNER_ * 2);   // 16 MB
  _Float16* h16  = (_Float16*)salloc((size_t)ROWS_X * DIM_ * 2);
  _Float16* q16  = (_Float16*)salloc((size_t)ROWS_X * INNER_ * 2);
  _Float16* k16  = (_Float16*)salloc((size_t)ROWS_X * INNER_ * 2);
  _Float16* v16  = (_Float16*)salloc((size_t)ROWS_X * INNER_ * 2);
  _Float16* O16  = (_Float16*)salloc((size_t)ROWS_X * INNER_ * 2);
  _Float16* hg16 = (_Float16*)salloc((size_t)ROWS_X * FF_INNER_ * 2);     // 32 MB
  _Float16* ctx16 = (_Float16*)salloc((size_t)ROWS_C * CTX_DIM_ * 2);
  float*    tbuf  = (float*)salloc((size_t)ROWS_X * RANK_ * 4);

  float* xcur = (float*)d_out;   // running residual lives directly in d_out

  auto cvt = [&](const float* s, _Float16* d, int n) {
    cvt_f16_kernel<<<(n + 255) / 256, 256, 0, stream>>>(s, d, n);
  };
  auto lora = [&](const _Float16* A, int M, int K, const float* dn, const float* al) {
    lora_down_kernel<<<(M * RANK_ + 255) / 256, 256, 0, stream>>>(A, M, K, dn, al, tbuf);
  };
  auto gemm16 = [&](const _Float16* A, int M, int K, int N, const float* lu, _Float16* dst) {
    dim3 grid(N / 64, (M + 127) / 128);
    gemm_wmma_kernel<0><<<grid, 256, 0, stream>>>(A, wbuf, M, K, N, nullptr, tbuf, lu,
                                                  dst, nullptr);
  };
  auto gemmres = [&](const _Float16* A, int M, int K, int N, const float* bias,
                     const float* lu) {
    dim3 grid(N / 64, (M + 127) / 128);
    gemm_wmma_kernel<1><<<grid, 256, 0, stream>>>(A, wbuf, M, K, N, bias, tbuf, lu,
                                                  nullptr, xcur);
  };
  auto attn_qkv = [&](const AttnP& a, const _Float16* hA, const _Float16* cA,
                      int Mc, int Kc) {
    cvt(a.Wq, wbuf, DIM_ * INNER_);
    lora(hA, ROWS_X, DIM_, a.qd, a.qa);
    gemm16(hA, ROWS_X, DIM_, INNER_, a.qu, q16);
    cvt(a.Wk, wbuf, Kc * INNER_);
    lora(cA, Mc, Kc, a.kd, a.ka);
    gemm16(cA, Mc, Kc, INNER_, a.ku, k16);
    cvt(a.Wv, wbuf, Kc * INNER_);
    lora(cA, Mc, Kc, a.vd, a.va);
    gemm16(cA, Mc, Kc, INNER_, a.vu, v16);
  };
  auto attn_out = [&](const AttnP& a) {
    cvt(a.Wo, wbuf, INNER_ * DIM_);
    lora(O16, ROWS_X, INNER_, a.od, a.oa);
    gemmres(O16, ROWS_X, INNER_, DIM_, a.bo, a.ou);
  };

  // x running copy
  hipMemcpyAsync(xcur, p.x, (size_t)ROWS_X * DIM_ * sizeof(float),
                 hipMemcpyDeviceToDevice, stream);

  // ---- self-attention ----
  ln_f16_kernel<<<ROWS_X, 256, 0, stream>>>(xcur, p.ln1_g, p.ln1_b, h16);
  attn_qkv(p.a1, h16, h16, ROWS_X, DIM_);
  attn_flash_kernel<<<dim3(NSEQ_ / 16, HEADS_, BATCH_), 32, 0, stream>>>(
      q16, k16, v16, O16, NSEQ_);
  attn_out(p.a1);

  // ---- cross-attention ----
  ln_f16_kernel<<<ROWS_X, 256, 0, stream>>>(xcur, p.ln2_g, p.ln2_b, h16);
  cvt(p.context, ctx16, ROWS_C * CTX_DIM_);
  attn_qkv(p.a2, h16, ctx16, ROWS_C, CTX_DIM_);
  attn_flash_kernel<<<dim3(NSEQ_ / 16, HEADS_, BATCH_), 32, 0, stream>>>(
      q16, k16, v16, O16, NCTX_);
  attn_out(p.a2);

  // ---- GEGLU feed-forward ----
  ln_f16_kernel<<<ROWS_X, 256, 0, stream>>>(xcur, p.ln3_g, p.ln3_b, h16);
  cvt(p.ff.Wp, wbuf, DIM_ * 2 * FF_INNER_);
  lora(h16, ROWS_X, DIM_, p.ff.pd, p.ff.pa);
  geglu_wmma_kernel<<<dim3(FF_INNER_ / 64, ROWS_X / 128), 256, 0, stream>>>(
      h16, wbuf, p.ff.bp, tbuf, p.ff.pu, hg16);
  cvt(p.ff.W2, wbuf, FF_INNER_ * DIM_);
  lora(hg16, ROWS_X, FF_INNER_, p.ff.fd, p.ff.fa);
  gemmres(hg16, ROWS_X, FF_INNER_, DIM_, p.ff.b2, p.ff.fu);
}
